// DenseFiLMResBlock_54159537603069
// MI455X (gfx1250) — matrix operand order repaired and checked by reading, not compiled, not run on hardware
//
#include <hip/hip_runtime.h>
#include <hip/hip_bf16.h>
#include <stdint.h>

// ---------------- problem constants ----------------
#define B_  32
#define S_  1024
#define D_  1024
#define E_  128
#define FE_ 512          // 4*E
#define M_  (B_ * S_)    // 32768
#define RCHUNK 64        // stats reduction chunks per batch

// ---------------- vector types ----------------
typedef __attribute__((ext_vector_type(16))) __bf16 v16bf;
typedef __attribute__((ext_vector_type(8)))  __bf16 v8bf;
typedef __attribute__((ext_vector_type(8)))  float  v8f;
typedef __attribute__((ext_vector_type(4)))  float  v4f;
typedef uint32_t u32x4 __attribute__((ext_vector_type(4)));
typedef uint32_t u32x8 __attribute__((ext_vector_type(8)));

// ---------------- GEMM tiling ----------------
#define TILE_M 128
#define TILE_N 128
#define TILE_K 32
#define KSTEPS (D_ / TILE_K)
#define LDSTR  40        // padded LDS row stride (bf16 elems): 80B, 16B-aligned rows

static __device__ __forceinline__ float silu_f(float a) {
    return a / (1.0f + __expf(-a));
}

// A-fragment (16x32 bf16, row-major tile in LDS): lane L (L<16) holds row M=L,
// K = {0..7, 16..23}; lanes 16..31 hold K = {8..15, 24..31}.
static __device__ __forceinline__ v16bf ldsFragA(const __bf16* rowbase, int khalf) {
    union { v16bf v; v8bf h[2]; } u;
    u.h[0] = *(const v8bf*)(rowbase + khalf);
    u.h[1] = *(const v8bf*)(rowbase + khalf + 16);
    return u.v;
}

// B-fragment from LDS holding Bt (N x K): lane L holds column N=L&15,
// K = {0..7, 16..23} (L<16) or {8..15, 24..31} (L>=16).
static __device__ __forceinline__ v16bf ldsFragB(const __bf16* rowbase, int koff) {
    union { v16bf v; v8bf h[2]; } u;
    u.h[0] = *(const v8bf*)(rowbase + koff);
    u.h[1] = *(const v8bf*)(rowbase + koff + 16);
    return u.v;
}

// Async copy of 16 bytes global -> LDS (CDNA5 GLOBAL_LOAD_ASYNC_TO_LDS_B128,
// GVS addressing: mem = SADDR64 + VADDR32). Tracked by ASYNCcnt.
static __device__ __forceinline__ void async_g2l_b128(uint32_t lds_byte_addr,
                                                      uint32_t g_byte_off,
                                                      uint64_t g_base) {
    asm volatile("global_load_async_to_lds_b128 %0, %1, %2"
                 :: "v"(lds_byte_addr), "v"(g_byte_off), "s"(g_base)
                 : "memory");
}
static __device__ __forceinline__ void wait_async0() {
    asm volatile("s_wait_asynccnt 0" ::: "memory");
}
static __device__ __forceinline__ void wait_tensor0() {
    asm volatile("s_wait_tensorcnt 0x0" ::: "memory");
}
static __device__ __forceinline__ uint32_t lds_addr_of(const void* p) {
    // generic shared address: high bits = shared aperture, low 32 = LDS offset
    return (uint32_t)(uintptr_t)p;
}

// One 16-row x 32-col bf16 tile of the transposed weight matrix via the
// Tensor Data Mover (inline asm: portable across both toolchains' builtin
// arities).  Global row stride = D_ elements; LDS rows padded to 80B via the
// TDM pad feature (16B pad after every 64B stored: pad_interval code 3 ->
// 16 DWORDs, pad_amount code 3 -> 4 DWORDs).  All inputs must be wave-uniform.
static __device__ __forceinline__ void tdm_load_tile(uint32_t lds_addr,
                                                     uint64_t g_addr_bytes) {
    const u32x4 g0 = {
        1u,                                                    // count=1 (valid), user D#
        lds_addr,                                              // lds_addr[31:0]
        (uint32_t)g_addr_bytes,                                // global_addr[31:0]
        ((uint32_t)(g_addr_bytes >> 32) & 0x01FFFFFFu) | 0x80000000u // [56:32] | type=2
    };
    const u32x8 g1 = {
        (1u << 16) | (1u << 20) | (3u << 22) | (3u << 25),     // data_size=2B, pad on
        (uint32_t)D_ << 16,                                    // tensor_dim0 = 1024
        0xFFFF0000u,                                           // tensor_dim1 = 65535
        (uint32_t)TILE_K << 16,                                // tile_dim0 = 32
        16u,                                                   // tile_dim1 = 16 rows/wave
        (uint32_t)D_,                                          // tensor_dim0_stride = 1024
        0u, 0u
    };
    const u32x4 gz = { 0u, 0u, 0u, 0u };
    asm volatile("tensor_load_to_lds %0, %1, %2, %3"
                 :: "s"(g0), "s"(g1), "s"(gz), "s"(gz)
                 : "memory");
}

// =====================================================================
// Kernel 1: DenseFiLM  (one block per batch, 512 threads)
// =====================================================================
__global__ void film_kernel(const float* __restrict__ t,
                            const float* __restrict__ W1, const float* __restrict__ b1,
                            const float* __restrict__ W2, const float* __restrict__ b2,
                            const float* __restrict__ Wsc, const float* __restrict__ bsc,
                            const float* __restrict__ Wsh, const float* __restrict__ bsh,
                            float* __restrict__ scale, float* __restrict__ shift)
{
    __shared__ float h0[E_];
    __shared__ float h1[FE_];
    __shared__ float h2[FE_];
    const int b   = blockIdx.x;
    const int tid = threadIdx.x;         // 0..511
    const float tv = t[b];

    if (tid < E_ / 2) {
        float fr = __expf(-logf(10000.0f) / 63.0f * (float)tid);
        float e  = 5000.0f * tv * fr;
        h0[tid]          = sinf(e);
        h0[tid + E_ / 2] = cosf(e);
    }
    __syncthreads();

    {   // h1 = silu(h0 @ W1 + b1)
        float acc = b1[tid];
        #pragma unroll 4
        for (int i = 0; i < E_; ++i) acc += h0[i] * W1[i * FE_ + tid];
        h1[tid] = silu_f(acc);
    }
    __syncthreads();

    {   // h2 = h1 @ W2 + b2
        float acc = b2[tid];
        #pragma unroll 4
        for (int i = 0; i < FE_; ++i) acc += h1[i] * W2[i * FE_ + tid];
        h2[tid] = acc;
    }
    __syncthreads();

    for (int d = tid; d < D_; d += FE_) {
        float asc = bsc[d], ash = bsh[d];
        #pragma unroll 4
        for (int i = 0; i < FE_; ++i) {
            float h = h2[i];
            asc += h * Wsc[i * D_ + d];
            ash += h * Wsh[i * D_ + d];
        }
        scale[b * D_ + d] = asc;
        shift[b * D_ + d] = ash;
    }
}

// =====================================================================
// Kernel 2/3: per-batch mean & rstd over S*D elements (two stages)
// =====================================================================
__global__ void ln_stats1(const float* __restrict__ src, float* __restrict__ partials)
{
    const int b = blockIdx.y, c = blockIdx.x, tid = threadIdx.x;   // 256 threads
    const float* p = src + (size_t)b * (S_ * D_) + (size_t)c * (S_ * D_ / RCHUNK);
    float s = 0.f, q = 0.f;
    #pragma unroll 4
    for (int i = 0; i < (S_ * D_ / RCHUNK / (256 * 4)); ++i) {   // 16 iters of b128
        v4f v = *(const v4f*)(p + (size_t)(i * 256 + tid) * 4);
        s += v[0] + v[1] + v[2] + v[3];
        q += v[0] * v[0] + v[1] * v[1] + v[2] * v[2] + v[3] * v[3];
    }
    __shared__ float ls[256], lq[256];
    ls[tid] = s; lq[tid] = q;
    __syncthreads();
    for (int o = 128; o > 0; o >>= 1) {
        if (tid < o) { ls[tid] += ls[tid + o]; lq[tid] += lq[tid + o]; }
        __syncthreads();
    }
    if (tid == 0) {
        partials[(b * RCHUNK + c) * 2 + 0] = ls[0];
        partials[(b * RCHUNK + c) * 2 + 1] = lq[0];
    }
}

__global__ void ln_stats2(const float* __restrict__ partials, float* __restrict__ stats)
{
    const int b = blockIdx.x, tid = threadIdx.x;   // 64 threads
    __shared__ float ls[RCHUNK], lq[RCHUNK];
    ls[tid] = partials[(b * RCHUNK + tid) * 2 + 0];
    lq[tid] = partials[(b * RCHUNK + tid) * 2 + 1];
    __syncthreads();
    for (int o = RCHUNK / 2; o > 0; o >>= 1) {
        if (tid < o) { ls[tid] += ls[tid + o]; lq[tid] += lq[tid + o]; }
        __syncthreads();
    }
    if (tid == 0) {
        const float n = (float)(S_ * D_);
        float mean = ls[0] / n;
        float var  = lq[0] / n - mean * mean;
        stats[b * 2 + 0] = mean;
        stats[b * 2 + 1] = rsqrtf(var + 1e-5f);
    }
}

// =====================================================================
// Kernel 4: fp32 W [K][N] -> transposed bf16 split Wt_hi/Wt_lo [N][K]
// =====================================================================
__global__ void prep_w(const float* __restrict__ W,
                       __bf16* __restrict__ Whi, __bf16* __restrict__ Wlo)
{
    const int e = blockIdx.x * 256 + threadIdx.x;   // 0 .. D_*D_-1
    const int k = e >> 10;
    const int n = e & (D_ - 1);
    float w = W[e];
    __bf16 h = (__bf16)w;
    __bf16 l = (__bf16)(w - (float)h);
    Whi[(size_t)n * D_ + k] = h;
    Wlo[(size_t)n * D_ + k] = l;
}

// =====================================================================
// Kernel 5: fused  out = silu(scale*LN(src)+shift) @ W  (+bias)(+residual)
// bf16x3 split-precision WMMA GEMM, double-buffered LDS.
//   hi weight tiles: TDM (tensor_load_to_lds, TENSORcnt), 16 rows per wave
//   lo weight tiles: GLOBAL_LOAD_ASYNC_TO_LDS_B128 (ASYNCcnt)
//   A tiles: fused LN+FiLM+silu transform (VALU) -> bf16 hi/lo in LDS
// grid = (N/128, M/128), block = 256 threads (8 waves, 4x2 wave grid,
// each wave computes 32x64 = 2x4 WMMA tiles).
// =====================================================================
__global__ void __launch_bounds__(256)
gemm_film(const float* __restrict__ src,   const float* __restrict__ gamma,
          const float* __restrict__ beta,  const float* __restrict__ scale,
          const float* __restrict__ shift, const float* __restrict__ stats,
          const __bf16* __restrict__ Wthi, const __bf16* __restrict__ Wtlo,
          const float* __restrict__ bias,  const float* __restrict__ residual,
          float* __restrict__ out)
{
    __shared__ __attribute__((aligned(16))) __bf16 lAhi[2][TILE_M * LDSTR];
    __shared__ __attribute__((aligned(16))) __bf16 lAlo[2][TILE_M * LDSTR];
    __shared__ __attribute__((aligned(16))) __bf16 lBhi[2][TILE_N * LDSTR];
    __shared__ __attribute__((aligned(16))) __bf16 lBlo[2][TILE_N * LDSTR];

    const int n0  = blockIdx.x * TILE_N;
    const int m0  = blockIdx.y * TILE_M;
    const int b   = m0 / S_;              // 128-row tiles never straddle a batch
    const int s0  = m0 & (S_ - 1);
    const int tid = threadIdx.x;
    const int lane = tid & 31;
    // wave id is uniform within a wave; readfirstlane makes that explicit so
    // the TDM descriptor (an "s"-constrained SGPR tuple) is provably uniform.
    const int wave  = __builtin_amdgcn_readfirstlane(tid >> 5);   // 0..7
    const int waveM = wave & 3;           // 4 along M (32 rows each)
    const int waveN = wave >> 2;          // 2 along N (64 cols each)

    const float mean = stats[b * 2 + 0];
    const float rstd = stats[b * 2 + 1];

    // A-tile loader assignment: each thread transforms 16 elements of one row
    const int arow = tid & 127;
    const int akh  = (tid >> 7) << 4;     // 0 or 16

    const float* xrow = src   + (size_t)(m0 + arow) * D_ + akh;
    const float* grow = gamma + (size_t)(s0 + arow) * D_ + akh;
    const float* brow = beta  + (size_t)(s0 + arow) * D_ + akh;
    const float* srow = scale + (size_t)b * D_ + akh;
    const float* hrow = shift + (size_t)b * D_ + akh;

    const uint64_t wHiBase = (uint64_t)(uintptr_t)Wthi;
    const uint64_t wLoBase = (uint64_t)(uintptr_t)Wtlo;

    // ---- weight-tile loader: hi via TDM (per-wave 16 rows), lo via async ----
    auto loadB = [&](int buf, int k0) {
        {
            const int r0 = wave << 4;   // this wave's 16 rows of the 128-row tile
            const uint64_t goff = (uint64_t)((((n0 + r0) << 10) + k0) * 2);
            tdm_load_tile(lds_addr_of(&lBhi[buf][r0 * LDSTR]), wHiBase + goff);
        }
        #pragma unroll
        for (int it = 0; it < 2; ++it) {
            const int c  = tid + it * 256;
            const int nr = c >> 2;
            const int kc = (c & 3) * 8;
            const uint32_t goff = (uint32_t)((((n0 + nr) << 10) + k0 + kc) * 2);
            async_g2l_b128(lds_addr_of(&lBlo[buf][nr * LDSTR + kc]), goff, wLoBase);
        }
    };

    auto waitTiles = [&]() {
        wait_tensor0();
        wait_async0();
    };

    // ---- A-tile loader: LN + FiLM + silu fused, bf16 hi/lo split ----
    auto loadA = [&](int buf, int k0) {
        union { v8bf h[2]; __bf16 e[16]; } uh, ul;
        #pragma unroll
        for (int j = 0; j < 4; ++j) {
            v4f xv = *(const v4f*)(xrow + k0 + 4 * j);
            v4f gv = *(const v4f*)(grow + k0 + 4 * j);
            v4f bv = *(const v4f*)(brow + k0 + 4 * j);
            v4f sv = *(const v4f*)(srow + k0 + 4 * j);
            v4f fv = *(const v4f*)(hrow + k0 + 4 * j);
            #pragma unroll
            for (int q = 0; q < 4; ++q) {
                float a = sv[q] * ((xv[q] - mean) * rstd * gv[q] + bv[q]) + fv[q];
                a = silu_f(a);
                __bf16 h = (__bf16)a;
                uh.e[4 * j + q] = h;
                ul.e[4 * j + q] = (__bf16)(a - (float)h);
            }
        }
        __bf16* dhi = &lAhi[buf][arow * LDSTR + akh];
        __bf16* dlo = &lAlo[buf][arow * LDSTR + akh];
        *(v8bf*)(dhi)     = uh.h[0];
        *(v8bf*)(dhi + 8) = uh.h[1];
        *(v8bf*)(dlo)     = ul.h[0];
        *(v8bf*)(dlo + 8) = ul.h[1];
    };

    v8f acc[2][4];
    #pragma unroll
    for (int i = 0; i < 2; ++i)
        #pragma unroll
        for (int j = 0; j < 4; ++j)
            #pragma unroll
            for (int q = 0; q < 8; ++q) acc[i][j][q] = 0.0f;

    // ---- prologue: fill buffer 0 ----
    loadB(0, 0);
    loadA(0, 0);
    waitTiles();
    __syncthreads();

    #pragma unroll 1
    for (int kt = 0; kt < KSTEPS; ++kt) {
        const int cur = kt & 1;
        const int k0  = kt * TILE_K;

        // issue next tile (TDM/async B + VALU-transformed A) before computing
        if (kt + 1 < KSTEPS) {
            loadB(cur ^ 1, k0 + TILE_K);
            loadA(cur ^ 1, k0 + TILE_K);
        }
        // prefetch A operands two tiles ahead into L2/WGP$
        if (kt + 2 < KSTEPS) {
            __builtin_prefetch(xrow + k0 + 2 * TILE_K, 0, 1);
            __builtin_prefetch(grow + k0 + 2 * TILE_K, 0, 1);
            __builtin_prefetch(brow + k0 + 2 * TILE_K, 0, 1);
        }

        // ---- fragments from current buffer ----
        const int rA    = lane & 15;
        const int khA   = (lane >> 4) * 8;
        const int koffB = (lane >> 4) * 8;
        v16bf aHi[2], aLo[2], bHi[4], bLo[4];
        #pragma unroll
        for (int i = 0; i < 2; ++i) {
            const __bf16* p = &lAhi[cur][(waveM * 32 + i * 16 + rA) * LDSTR];
            const __bf16* q = &lAlo[cur][(waveM * 32 + i * 16 + rA) * LDSTR];
            aHi[i] = ldsFragA(p, khA);
            aLo[i] = ldsFragA(q, khA);
        }
        #pragma unroll
        for (int j = 0; j < 4; ++j) {
            const __bf16* p = &lBhi[cur][(waveN * 64 + j * 16 + rA) * LDSTR];
            const __bf16* q = &lBlo[cur][(waveN * 64 + j * 16 + rA) * LDSTR];
            bHi[j] = ldsFragB(p, koffB);
            bLo[j] = ldsFragB(q, koffB);
        }
        // ---- 2x4 tiles x 3 split products = 24 WMMA per wave per k-step ----
        #pragma unroll
        for (int i = 0; i < 2; ++i) {
            #pragma unroll
            for (int j = 0; j < 4; ++j) {
                acc[i][j] = __builtin_amdgcn_wmma_f32_16x16x32_bf16(
                    false, aHi[i], false, bHi[j], (short)0, acc[i][j], false, false);
                acc[i][j] = __builtin_amdgcn_wmma_f32_16x16x32_bf16(
                    false, aHi[i], false, bLo[j], (short)0, acc[i][j], false, false);
                acc[i][j] = __builtin_amdgcn_wmma_f32_16x16x32_bf16(
                    false, aLo[i], false, bHi[j], (short)0, acc[i][j], false, false);
            }
        }
        // all DMA writes into the next buffer must land before the barrier
        waitTiles();
        __syncthreads();
    }

    // ---- epilogue: C layout = lane(0-15): N=lane, VGPR r -> M=r; lanes 16-31: M=8+r
    #pragma unroll
    for (int i = 0; i < 2; ++i) {
        #pragma unroll
        for (int j = 0; j < 4; ++j) {
            const int gn = n0 + waveN * 64 + j * 16 + (lane & 15);
            const int gm = m0 + waveM * 32 + i * 16 + (lane >> 4) * 8;
            const float bs = bias[gn];
            #pragma unroll
            for (int r = 0; r < 8; ++r) {
                const size_t off = (size_t)(gm + r) * D_ + gn;
                float v = acc[i][j][r] + bs;
                if (residual) v += residual[off];
                out[off] = v;
            }
        }
    }
}

// =====================================================================
// Host launcher
// =====================================================================
extern "C" void kernel_launch(void* const* d_in, const int* in_sizes, int n_in,
                              void* d_out, int out_size, void* d_ws, size_t ws_size,
                              hipStream_t stream)
{
    (void)in_sizes; (void)n_in; (void)out_size; (void)ws_size;
    const float* x     = (const float*)d_in[0];
    const float* t     = (const float*)d_in[1];
    const float* W1    = (const float*)d_in[2];
    const float* b1    = (const float*)d_in[3];
    const float* W2    = (const float*)d_in[4];
    const float* b2    = (const float*)d_in[5];
    const float* Wsc   = (const float*)d_in[6];
    const float* bsc   = (const float*)d_in[7];
    const float* Wsh   = (const float*)d_in[8];
    const float* bsh   = (const float*)d_in[9];
    const float* gamma = (const float*)d_in[10];
    const float* beta  = (const float*)d_in[11];
    const float* W_in  = (const float*)d_in[12];
    const float* b_in  = (const float*)d_in[13];
    const float* W_out = (const float*)d_in[14];
    const float* b_out = (const float*)d_in[15];
    float* out = (float*)d_out;

    size_t off = 0;
    auto take = [&](size_t bytes) -> char* {
        char* p = (char*)d_ws + off;
        off += (bytes + 255) & ~(size_t)255;
        return p;
    };
    float*  scale    = (float*)take((size_t)B_ * D_ * sizeof(float));
    float*  shift    = (float*)take((size_t)B_ * D_ * sizeof(float));
    float*  statsX   = (float*)take((size_t)B_ * 2 * sizeof(float));
    float*  statsY   = (float*)take((size_t)B_ * 2 * sizeof(float));
    float*  partials = (float*)take((size_t)B_ * RCHUNK * 2 * sizeof(float));
    __bf16* Wthi1    = (__bf16*)take((size_t)D_ * D_ * sizeof(__bf16));
    __bf16* Wtlo1    = (__bf16*)take((size_t)D_ * D_ * sizeof(__bf16));
    __bf16* Wthi2    = (__bf16*)take((size_t)D_ * D_ * sizeof(__bf16));
    __bf16* Wtlo2    = (__bf16*)take((size_t)D_ * D_ * sizeof(__bf16));
    float*  y1       = (float*)take((size_t)M_ * D_ * sizeof(float));

    // 1. FiLM scale/shift
    film_kernel<<<B_, FE_, 0, stream>>>(t, W1, b1, W2, b2, Wsc, bsc, Wsh, bsh,
                                        scale, shift);
    // 2. LN stats of x
    ln_stats1<<<dim3(RCHUNK, B_), 256, 0, stream>>>(x, partials);
    ln_stats2<<<B_, RCHUNK, 0, stream>>>(partials, statsX);
    // 3. weight prep (both layers)
    prep_w<<<(D_ * D_) / 256, 256, 0, stream>>>(W_in,  Wthi1, Wtlo1);
    prep_w<<<(D_ * D_) / 256, 256, 0, stream>>>(W_out, Wthi2, Wtlo2);
    // 4. y1 = silu(scale*LN(x)+shift) @ W_in + b_in
    gemm_film<<<dim3(D_ / TILE_N, M_ / TILE_M), 256, 0, stream>>>(
        x, gamma, beta, scale, shift, statsX, Wthi1, Wtlo1, b_in, nullptr, y1);
    // 5. LN stats of y1
    ln_stats1<<<dim3(RCHUNK, B_), 256, 0, stream>>>(y1, partials);
    ln_stats2<<<B_, RCHUNK, 0, stream>>>(partials, statsY);
    // 6. out = silu(scale*LN(y1)+shift) @ W_out + b_out + x
    gemm_film<<<dim3(D_ / TILE_N, M_ / TILE_M), 256, 0, stream>>>(
        y1, gamma, beta, scale, shift, statsY, Wthi2, Wtlo2, b_out, x, out);
}
